// DynamicModel_88519275970731
// MI455X (gfx1250) — matrix operand order repaired and checked by reading, not compiled, not run on hardware
//
#include <hip/hip_runtime.h>

// ---------------------------------------------------------------------------
// CDNA5 / gfx1250 vector types for WMMA (wave32)
// ---------------------------------------------------------------------------
typedef __attribute__((ext_vector_type(16))) __bf16 v16bf;
typedef __attribute__((ext_vector_type(8)))  __bf16 v8bf;
typedef __attribute__((ext_vector_type(8)))  float  v8f;

#define BATCH 16
#define HH 256
#define WW 256
#define C0 32          // conv0 out channels
#define C1 64          // conv1 out channels
#define KDIM 512       // MLP layer-3 K
#define TOTAL_W 19296  // MLP layer-3 width
#define NTILES 1206    // 19296 / 16

__device__ __forceinline__ v16bf cat16(v8bf lo, v8bf hi) {
  return __builtin_shufflevector(lo, hi, 0,1,2,3,4,5,6,7,8,9,10,11,12,13,14,15);
}

// ---------------------------------------------------------------------------
// 1) MLP layers 0+1 fused (tiny): h = relu(relu(xm@w0+b0)@w1+b1) -> bf16 [16][512]
// ---------------------------------------------------------------------------
__global__ __launch_bounds__(256) void mlp12_kernel(
    const float* __restrict__ xm, const float* __restrict__ w0,
    const float* __restrict__ b0, const float* __restrict__ w1,
    const float* __restrict__ b1, __bf16* __restrict__ gh) {
  __shared__ float sx[16 * 16];
  __shared__ float sh0[16 * 256];
  const int t = threadIdx.x;
  sx[t] = xm[t];               // 256 == 16*16
  __syncthreads();
  // layer 0: thread t owns column t for all 16 rows
  for (int bb = 0; bb < 16; ++bb) {
    float a = b0[t];
    #pragma unroll
    for (int k = 0; k < 16; ++k) a += sx[bb * 16 + k] * w0[k * 256 + t];
    sh0[bb * 256 + t] = fmaxf(a, 0.f);
  }
  __syncthreads();
  // layer 1: thread t owns columns t and t+256
  for (int bb = 0; bb < 16; ++bb) {
    #pragma unroll
    for (int jj = 0; jj < 2; ++jj) {
      const int j = t + jj * 256;
      float a = b1[j];
      for (int k = 0; k < 256; ++k) a += sh0[bb * 256 + k] * w1[k * 512 + j];
      gh[bb * 512 + j] = (__bf16)fmaxf(a, 0.f);
    }
  }
}

// ---------------------------------------------------------------------------
// 2) w2 [512][19296] f32  ->  w2t [19296][512] bf16 (transposed, K-contiguous)
//    64k x 32n tile bounced through LDS so BOTH global sides are coalesced:
//    float4 row-segment reads -> ds_load_b128 + global_store_b128 k-octets.
//    Grid 8 x 603 covers 512 x 19296 exactly.
// ---------------------------------------------------------------------------
__global__ __launch_bounds__(256) void cvt_w2t_kernel(
    const float* __restrict__ w2, __bf16* __restrict__ w2t) {
  __shared__ __bf16 sT[32][72];       // [n_local][k_local], padded rows (144 B)
  const int kt = blockIdx.x;          // k tile (64 wide)
  const int nt = blockIdx.y;          // n tile (32 wide)
  const int t  = threadIdx.x;
  #pragma unroll
  for (int it = 0; it < 2; ++it) {
    const int slot = t + it * 256;    // 0..511 = 64 k rows x 8 n-quads
    const int kl = slot >> 3;
    const int nq = (slot & 7) * 4;
    const float4 v =
        *(const float4*)(w2 + (size_t)(kt * 64 + kl) * TOTAL_W + nt * 32 + nq);
    sT[nq + 0][kl] = (__bf16)v.x;
    sT[nq + 1][kl] = (__bf16)v.y;
    sT[nq + 2][kl] = (__bf16)v.z;
    sT[nq + 3][kl] = (__bf16)v.w;
  }
  __syncthreads();
  const int nl = t >> 3;              // 0..31
  const int ko = (t & 7) * 8;         // 0..56
  v8bf o = *(const v8bf*)(&sT[nl][ko]);
  *(v8bf*)(w2t + (size_t)(nt * 32 + nl) * KDIM + kt * 64 + ko) = o;
}

// ---------------------------------------------------------------------------
// 3) MLP layer 3 via WMMA bf16: flat[16][19296] = relu(h@w2 + b2)
//    One wave per 16-wide N tile; K = 512 = 16 fully-unrolled WMMA steps.
//    With w2t K-contiguous, BOTH fragments are two global_load_b128 per lane.
//    Barrier-free, wave-uniform exit -> EXEC all-ones at every WMMA.
// ---------------------------------------------------------------------------
__global__ __launch_bounds__(128) void mlp3_wmma_kernel(
    const __bf16* __restrict__ gh, const __bf16* __restrict__ w2t,
    const float* __restrict__ b2, float* __restrict__ flat) {
  const int wave  = threadIdx.x >> 5;
  const int lane  = threadIdx.x & 31;
  const int ntile = blockIdx.x * 4 + wave;
  if (ntile >= NTILES) return;          // wave-uniform: whole wave exits together
  const int mrow = lane & 15;
  const int hi   = lane >> 4;           // 0: low K half, 1: high K half
  const int n    = ntile * 16 + mrow;
  // A (contiguous k order, same as B): row M=lane&15, K = 16*hi + {0..15}
  const __bf16* hp = gh + (size_t)mrow * KDIM + hi * 16;
  // B: col N=lane&15, K = 16*hi + {0..15}, contiguous in w2t[n][k]
  const __bf16* bp = w2t + (size_t)n * KDIM + hi * 16;
  v8f acc = {};
  #pragma unroll
  for (int kc = 0; kc < 16; ++kc) {
    v16bf afrag = cat16(*(const v8bf*)(hp + kc * 32),
                        *(const v8bf*)(hp + kc * 32 + 8));
    v16bf bfrag = cat16(*(const v8bf*)(bp + kc * 32),
                        *(const v8bf*)(bp + kc * 32 + 8));
    acc = __builtin_amdgcn_wmma_f32_16x16x32_bf16(
        false, afrag, false, bfrag, (short)0, acc, false, false);
  }
  const float bias = b2[n];
  #pragma unroll
  for (int r = 0; r < 8; ++r) {
    const int m = hi * 8 + r;           // C/D layout: VGPR r, halves split at M=8
    flat[(size_t)m * TOTAL_W + n] = fmaxf(acc[r] + bias, 0.f);
  }
}

// ---------------------------------------------------------------------------
// 4) Repack generated filters into WMMA/B-friendly bf16 tensors
//    conv0: [b][tap][oc]        <- flat[b][oc*9 + tap]
//    conv1: [b][tap][oc][ic]    <- flat[b][288 + oc*288 + ic*9 + tap]
//    conv2: [b][tap][ic]        <- flat[b][18720 + ic*9 + tap]
// ---------------------------------------------------------------------------
__global__ __launch_bounds__(256) void repack0_kernel(
    const float* __restrict__ flat, __bf16* __restrict__ w0c) {
  const int i = blockIdx.x * 256 + threadIdx.x;
  if (i >= BATCH * 9 * C0) return;
  const int oc = i & 31, tap = (i >> 5) % 9, b = i / (9 * 32);
  w0c[i] = (__bf16)flat[(size_t)b * TOTAL_W + oc * 9 + tap];
}
__global__ __launch_bounds__(256) void repack1_kernel(
    const float* __restrict__ flat, __bf16* __restrict__ w1c) {
  const int i = blockIdx.x * 256 + threadIdx.x;   // 294912 total, exact grid
  const int ic = i & 31, oc = (i >> 5) & 63;
  const int tap = (i >> 11) % 9, b = (i >> 11) / 9;
  w1c[i] = (__bf16)flat[(size_t)b * TOTAL_W + 288 + oc * 288 + ic * 9 + tap];
}
__global__ __launch_bounds__(256) void repack2_kernel(
    const float* __restrict__ flat, __bf16* __restrict__ w2c) {
  const int i = blockIdx.x * 256 + threadIdx.x;
  if (i >= BATCH * 9 * C1) return;
  const int ic = i & 63, tap = (i >> 6) % 9, b = i / (9 * 64);
  w2c[i] = (__bf16)flat[(size_t)b * TOTAL_W + 18720 + ic * 9 + tap];
}

// ---------------------------------------------------------------------------
// 5) conv0 (1 -> 32), SAME: memory-bound VALU; writes bf16 NHWC act0
// ---------------------------------------------------------------------------
__global__ __launch_bounds__(256) void conv0_kernel(
    const float* __restrict__ x, const __bf16* __restrict__ w0c,
    __bf16* __restrict__ act0) {
  const int idx = blockIdx.x * 256 + threadIdx.x;    // (b*256 + y)*256 + px
  const int px = idx & 255, y = (idx >> 8) & 255, b = idx >> 16;
  float acc[C0];
  #pragma unroll
  for (int i = 0; i < C0; ++i) acc[i] = 0.f;
  #pragma unroll
  for (int tap = 0; tap < 9; ++tap) {
    const int yy = y + tap / 3 - 1, xx = px + tap % 3 - 1;
    if (yy < 0 || yy >= HH || xx < 0 || xx >= WW) continue;
    const float v = x[((size_t)b * HH + yy) * WW + xx];
    const __bf16* wp = w0c + ((size_t)b * 9 + tap) * C0;
    #pragma unroll
    for (int oc = 0; oc < C0; ++oc) acc[oc] += v * (float)wp[oc];
  }
  __bf16* o = act0 + (size_t)idx * C0;
  #pragma unroll
  for (int j = 0; j < 4; ++j) {
    v8bf pk;
    #pragma unroll
    for (int k = 0; k < 8; ++k) pk[k] = (__bf16)acc[j * 8 + k];
    *(v8bf*)(o + j * 8) = pk;
  }
}

// ---------------------------------------------------------------------------
// 6) conv1 (32 -> 64) via WMMA implicit GEMM  -- the 95%-of-FLOPs kernel.
//    Block = 8 waves = 32 pixels x 64 out-channels; wave tile = 16px x 16oc.
//    The 3-row x 34-px x 32-ch input halo (6.5 KB) is staged into LDS once
//    (zero-padded), so the 9-tap loop is branch-free:
//    2x ds_load_b128 (A) + 2x global_load_b128 (B, L0-resident) + v_wmma.
// ---------------------------------------------------------------------------
__global__ __launch_bounds__(256) void conv1_wmma_kernel(
    const __bf16* __restrict__ act0, const __bf16* __restrict__ w1c,
    __bf16* __restrict__ act1) {
  __shared__ __bf16 sA[3 * 34 * C0];      // 6528 B halo tile
  const int xt = blockIdx.x;              // 8 tiles of 32 px
  const int y  = blockIdx.y;              // row
  const int b  = blockIdx.z;              // sample
  const int t  = threadIdx.x;
  const int wave = t >> 5;
  const int lane = t & 31;
  const int psub = wave & 1;              // pixel subtile (16 px)
  const int ntile = wave >> 1;            // oc subtile (16 oc)
  const int mrow = lane & 15;
  const int hi   = lane >> 4;
  const int x0 = xt * 32;

  // Stage halo: 408 16-byte chunks, zero pad outside the image (SAME padding).
  for (int c = t; c < 3 * 34 * 4; c += 256) {
    const int r = c / 136, rem = c - r * 136;   // 136 = 34 px * 4 chunks
    const int p = rem >> 2, q = rem & 3;
    const int yy = y + r - 1, xx = x0 - 1 + p;
    v8bf v = {};
    if (yy >= 0 && yy < HH && xx >= 0 && xx < WW)
      v = *(const v8bf*)(act0 + (((size_t)b * HH + yy) * WW + xx) * C0 + q * 8);
    *(v8bf*)(&sA[(size_t)c * 8]) = v;
  }
  __syncthreads();

  const int n = ntile * 16 + mrow;        // B col -> out channel
  v8f acc = {};
  #pragma unroll
  for (int tap = 0; tap < 9; ++tap) {
    const int ky = tap / 3, kx = tap % 3;
    // A fragment from LDS: pixel p = psub*16 + mrow + kx of halo row ky
    const int p = psub * 16 + mrow + kx;
    const __bf16* ap = &sA[(ky * 34 + p) * C0 + hi * 16];
    v16bf afrag = cat16(*(const v8bf*)ap, *(const v8bf*)(ap + 8));
    // B fragment: w1c[b][tap][n][ic], ic = 16*hi + 0..15 (two contiguous b128s)
    const __bf16* wp = w1c + (((size_t)b * 9 + tap) * C1 + n) * C0 + hi * 16;
    v16bf bfrag = cat16(*(const v8bf*)wp, *(const v8bf*)(wp + 8));
    acc = __builtin_amdgcn_wmma_f32_16x16x32_bf16(
        false, afrag, false, bfrag, (short)0, acc, false, false);
  }
  // C/D layout: VGPR r -> pixel m = 8*hi + r, channel = ntile*16 + (lane&15)
  #pragma unroll
  for (int r = 0; r < 8; ++r) {
    const int m = hi * 8 + r;
    const int pxo = x0 + psub * 16 + m;
    act1[(((size_t)b * HH + y) * WW + pxo) * C1 + n] = (__bf16)acc[r];
  }
}

// ---------------------------------------------------------------------------
// 7) conv2 (64 -> 1): memory-bound reduction, per-sample weights in LDS
// ---------------------------------------------------------------------------
__global__ __launch_bounds__(256) void conv2_kernel(
    const __bf16* __restrict__ act1, const __bf16* __restrict__ w2c,
    float* __restrict__ out) {
  __shared__ float wsh[9 * C1];
  const int y = blockIdx.x, b = blockIdx.y, px = threadIdx.x;
  for (int i = threadIdx.x; i < 9 * C1; i += 256)
    wsh[i] = (float)w2c[(size_t)b * 9 * C1 + i];
  __syncthreads();
  float acc = 0.f;
  #pragma unroll
  for (int tap = 0; tap < 9; ++tap) {
    const int yy = y + tap / 3 - 1, xx = px + tap % 3 - 1;
    if (yy < 0 || yy >= HH || xx < 0 || xx >= WW) continue;
    const __bf16* p = act1 + (((size_t)b * HH + yy) * WW + xx) * C1;
    #pragma unroll
    for (int j = 0; j < 8; ++j) {
      v8bf v = *(const v8bf*)(p + j * 8);
      #pragma unroll
      for (int k = 0; k < 8; ++k) acc += (float)v[k] * wsh[tap * C1 + j * 8 + k];
    }
  }
  out[((size_t)b * HH + y) * WW + px] = acc;
}

// ---------------------------------------------------------------------------
// Launch
// ---------------------------------------------------------------------------
extern "C" void kernel_launch(void* const* d_in, const int* in_sizes, int n_in,
                              void* d_out, int out_size, void* d_ws, size_t ws_size,
                              hipStream_t stream) {
  const float* xm = (const float*)d_in[0];
  const float* x  = (const float*)d_in[1];
  const float* w0 = (const float*)d_in[2];
  const float* b0 = (const float*)d_in[3];
  const float* w1 = (const float*)d_in[4];
  const float* b1 = (const float*)d_in[5];
  const float* w2 = (const float*)d_in[6];
  const float* b2 = (const float*)d_in[7];
  float* out = (float*)d_out;

  char* ws = (char*)d_ws;
  size_t off = 0;
  auto carve = [&](size_t bytes) -> char* {
    char* p = ws + off;
    off += (bytes + 255) & ~(size_t)255;
    return p;
  };
  __bf16* gh   = (__bf16*)carve((size_t)16 * KDIM * 2);                 // h (bf16)
  float*  flat = (float*) carve((size_t)16 * TOTAL_W * 4);              // MLP out
  __bf16* w2t  = (__bf16*)carve((size_t)TOTAL_W * KDIM * 2);            // w2^T bf16
  __bf16* w0c  = (__bf16*)carve((size_t)BATCH * 9 * C0 * 2);
  __bf16* w1c  = (__bf16*)carve((size_t)BATCH * 9 * C1 * C0 * 2);
  __bf16* w2c  = (__bf16*)carve((size_t)BATCH * 9 * C1 * 2);
  __bf16* act0 = (__bf16*)carve((size_t)BATCH * HH * WW * C0 * 2);      // 64 MB
  __bf16* act1 = (__bf16*)carve((size_t)BATCH * HH * WW * C1 * 2);      // 128 MB

  cvt_w2t_kernel<<<dim3(KDIM / 64, TOTAL_W / 32), 256, 0, stream>>>(w2, w2t);
  mlp12_kernel<<<1, 256, 0, stream>>>(xm, w0, b0, w1, b1, gh);
  mlp3_wmma_kernel<<<(NTILES + 3) / 4, 128, 0, stream>>>(gh, w2t, b2, flat);
  repack0_kernel<<<(BATCH * 9 * C0 + 255) / 256, 256, 0, stream>>>(flat, w0c);
  repack1_kernel<<<(BATCH * 9 * C1 * C0) / 256, 256, 0, stream>>>(flat, w1c);
  repack2_kernel<<<(BATCH * 9 * C1 + 255) / 256, 256, 0, stream>>>(flat, w2c);
  conv0_kernel<<<(BATCH * HH * WW) / 256, 256, 0, stream>>>(x, w0c, act0);
  conv1_wmma_kernel<<<dim3(WW / 32, HH, BATCH), 256, 0, stream>>>(act0, w1c, act1);
  conv2_kernel<<<dim3(HH, BATCH), 256, 0, stream>>>(act1, w2c, out);
}
